// kuaishouEBR_73675868996394
// MI455X (gfx1250) — matrix operand order repaired and checked by reading, not compiled, not run on hardware
//
#include <hip/hip_runtime.h>
#include <hip/hip_bf16.h>

// ---------------------------------------------------------------------------
// Problem constants (match reference): B,S,D,H,K(clusters),R,N
// ---------------------------------------------------------------------------
#define BB 64
#define SS 50
#define DD 128
#define HH 512
#define KC 16
#define RR 50
#define NN 100000

typedef unsigned int   u32;
typedef unsigned short u16;

typedef __attribute__((ext_vector_type(16))) __bf16 v16bf;
typedef __attribute__((ext_vector_type(8)))  float  v8f;
typedef __attribute__((ext_vector_type(4)))  u32    v4u;

union BfFrag { v4u q[2]; v16bf v; };

// ---------------------------------------------------------------------------
// bf16 helpers (raw-bit ushort storage; RNE conversion)
// ---------------------------------------------------------------------------
__device__ __forceinline__ u16 f32_to_bf16(float f) {
    u32 u = __float_as_uint(f);
    u += 0x7FFFu + ((u >> 16) & 1u);      // round-to-nearest-even
    return (u16)(u >> 16);
}
__device__ __forceinline__ float bf16_to_f32(u16 h) {
    return __uint_as_float(((u32)h) << 16);
}

// ---------------------------------------------------------------------------
// Elementwise fp32 -> bf16 convert (row-major preserved; used for A matrices)
// ---------------------------------------------------------------------------
__global__ void cvt_f32_bf16(const float* __restrict__ x, u16* __restrict__ y, int n) {
    int i = blockIdx.x * blockDim.x + threadIdx.x;
    int stride = gridDim.x * blockDim.x;
    for (; i < n; i += stride) y[i] = f32_to_bf16(x[i]);
}

// ---------------------------------------------------------------------------
// Transpose-convert: W[K,N] fp32 (row-major) -> Wt[N,K] bf16 (row-major).
// 32x32 LDS tile, coalesced reads and writes. K is a multiple of 32.
// ---------------------------------------------------------------------------
__global__ __launch_bounds__(256) void cvt_transpose_f32_bf16(
    const float* __restrict__ W, u16* __restrict__ Wt, int K, int N) {
    __shared__ float tile[32][33];
    const int tx = threadIdx.x;           // 0..31
    const int ty = threadIdx.y;           // 0..7
    const int nt = blockIdx.x * 32;
    const int kt = blockIdx.y * 32;
#pragma unroll
    for (int i = 0; i < 4; ++i) {
        int k = kt + ty + 8 * i;          // always < K (K multiple of 32)
        int n = nt + tx;
        tile[ty + 8 * i][tx] = (n < N) ? W[(size_t)k * N + n] : 0.0f;
    }
    __syncthreads();
#pragma unroll
    for (int i = 0; i < 4; ++i) {
        int n = nt + ty + 8 * i;
        int k = kt + tx;
        if (n < N) Wt[(size_t)n * K + k] = f32_to_bf16(tile[tx][ty + 8 * i]);
    }
}

// ---------------------------------------------------------------------------
// Build prompted user history: [B, S*2D] bf16.
// ---------------------------------------------------------------------------
__global__ void build_prompted(const float* __restrict__ user,
                               const float* __restrict__ prompt_emb,
                               const int* __restrict__ hist,
                               u16* __restrict__ out) {
    int b = blockIdx.x;
    for (int c = threadIdx.x; c < SS * 2 * DD; c += blockDim.x) {
        int s = c >> 8;          // / 256
        int j = c & 255;
        float v;
        if (j < DD) v = user[((size_t)b * SS + s) * DD + j];
        else        v = prompt_emb[(size_t)hist[b * SS + s] * DD + (j - DD)];
        out[(size_t)b * (SS * 2 * DD) + c] = f32_to_bf16(v);
    }
}

// ---------------------------------------------------------------------------
// bf16 WMMA GEMM, compile-time specialized.
//   C[M,N] = act(A[M,Kd] * Bt^T + bias), optional rowScale/colScale epilogue.
//   Bt is ALWAYS [N,Kd] row-major (weights pre-transposed; itB naturally so).
// Tile 128(M) x 64(N) x 32(K); 256 thr = 8 waves; wave -> 16 rows x 64 cols
// (4 accumulators), v_wmma_f32_16x16x32_bf16.
// OOB staging rows/cols are CLAMPED (not zero-filled): a C element depends
// only on its own A row / B column, and the epilogue guards stores, so OOB
// tiles compute harmless garbage with zero branch overhead in the hot loop.
// LDS rows padded to 40 ushorts (80B): 16B-aligned b128 ds ops, conflict-free.
// ---------------------------------------------------------------------------
#define BM 128
#define BN 64
#define BK 32
#define LPAD 40

template<bool OUTBF16, bool SCALED, bool HASBIAS, bool RELU>
__global__ __launch_bounds__(256) void wmma_gemm(
    const u16* __restrict__ A, const u16* __restrict__ Bt,
    const float* __restrict__ bias,
    const float* __restrict__ rowScale, const float* __restrict__ colScale,
    void* __restrict__ Cout,
    int M, int N, int Kd)
{
    __shared__ u16 As[BM][LPAD];
    __shared__ u16 Bs[BN][LPAD];

    const int tid  = threadIdx.x;
    const int wave = tid >> 5;
    const int lane = tid & 31;
    const int m_base = blockIdx.y * BM;
    const int n_base = blockIdx.x * BN;
    const int mw   = m_base + wave * 16;

    v8f acc[4];
#pragma unroll
    for (int s = 0; s < 4; ++s)
#pragma unroll
        for (int j = 0; j < 8; ++j) acc[s][j] = 0.0f;

    // staging thread mappings (fixed per thread); OOB rows clamped.
    const int aRow = tid >> 1, aHalf = tid & 1;          // A: row, 16-elem half
    const int bRow = tid >> 2, bQ    = tid & 3;          // B: row, 8-elem quarter
    int gm = m_base + aRow; if (gm >= M) gm = M - 1;
    int gn = n_base + bRow; if (gn >= N) gn = N - 1;
    const u16* aSrc = A  + (size_t)gm * Kd + aHalf * 16;
    const u16* bSrc = Bt + (size_t)gn * Kd + bQ * 8;
    v4u* aDst = (v4u*)&As[aRow][aHalf * 16];
    v4u* bDst = (v4u*)&Bs[bRow][bQ * 8];

    for (int k0 = 0; k0 < Kd; k0 += BK) {
        // ---- stage A tile (128x32 bf16): 2 x b128 per thread, unconditional
        {
            const v4u* src = (const v4u*)(aSrc + k0);
            v4u t0 = src[0];
            v4u t1 = src[1];
            aDst[0] = t0;
            aDst[1] = t1;
        }
        // ---- stage B tile: Bs[n][k] straight from Bt[N,Kd] rows
        {
            v4u t = *(const v4u*)(bSrc + k0);
            *bDst = t;
        }
        if (k0 + BK < Kd) {                // wave-uniform scalar branch
            __builtin_prefetch(aSrc + k0 + BK, 0, 1);
            __builtin_prefetch(bSrc + k0 + BK, 0, 1);
        }
        __syncthreads();

        // ---- fragments + WMMA
        {
            const int mloc = wave * 16 + (lane & 15);
            const int h    = lane >> 4;
            BfFrag af;
            // A 16-bit layout: lanes 0-15 hold k = 0..7 (V0-3) & 16..23 (V4-7);
            // lanes 16-31 hold k = 8..15 & 24..31  -> two contiguous 16B runs.
            af.q[0] = *(const v4u*)&As[mloc][h * 8];
            af.q[1] = *(const v4u*)&As[mloc][16 + h * 8];
#pragma unroll
            for (int sub = 0; sub < 4; ++sub) {
                const int nloc = sub * 16 + (lane & 15);
                BfFrag bfg;
                // B layout: lane -> col n = lane%16, k = h*8 + i (i<8), 16 + h*8 + (i-8) (i>=8)
                bfg.q[0] = *(const v4u*)&Bs[nloc][h * 8];
                bfg.q[1] = *(const v4u*)&Bs[nloc][16 + h * 8];
                acc[sub] = __builtin_amdgcn_wmma_f32_16x16x32_bf16(
                    false, af.v, false, bfg.v, (short)0, acc[sub], false, false);
            }
        }
        __syncthreads();
    }

    // ---- epilogue: lane -> col = lane%16, VGPR j -> row = 8*(lane/16)+j
    {
        const int h  = lane >> 4;
        const int nc = lane & 15;
        float* Cf = (float*)Cout;
        u16*   Cb = (u16*)Cout;
#pragma unroll
        for (int sub = 0; sub < 4; ++sub) {
            int n = n_base + sub * 16 + nc;
            if (n >= N) continue;
            float cs = SCALED  ? colScale[n] : 1.0f;
            float bi = HASBIAS ? bias[n]     : 0.0f;
#pragma unroll
            for (int j = 0; j < 8; ++j) {
                int m = mw + 8 * h + j;
                if (m >= M) continue;
                float v = acc[sub][j];
                if (SCALED) v *= rowScale[m] * cs;
                if (HASBIAS) v += bi;
                if (RELU) v = fmaxf(v, 0.0f);
                if (OUTBF16) Cb[(size_t)m * N + n] = f32_to_bf16(v);
                else         Cf[(size_t)m * N + n] = v;
            }
        }
    }
}

// ---------------------------------------------------------------------------
// Per-row inverse L2 norm of a bf16 [rows, 512] matrix: one wave per row.
// ---------------------------------------------------------------------------
__global__ void row_inv_norm(const u16* __restrict__ X, float* __restrict__ inv, int rows) {
    int wave = threadIdx.x >> 5, lane = threadIdx.x & 31;
    int row = blockIdx.x * 8 + wave;
    if (row >= rows) return;
    float s = 0.0f;
    for (int c = lane; c < HH; c += 32) {
        float v = bf16_to_f32(X[(size_t)row * HH + c]);
        s += v * v;
    }
#pragma unroll
    for (int off = 16; off > 0; off >>= 1) s += __shfl_down(s, off);
    if (lane == 0) inv[row] = 1.0f / fmaxf(sqrtf(s), 1e-8f);
}

// ---------------------------------------------------------------------------
// Deterministic per-(b,k) top-R: ballot-compacted cluster sims in LDS,
// then R argmax rounds with index tie-break.
// ---------------------------------------------------------------------------
#define TOPK_CAP 8192

__global__ void topk_per_cluster(const float* __restrict__ sims,
                                 const int* __restrict__ labels,
                                 int* __restrict__ outIdx) {
    extern __shared__ char smem[];
    float* vals = (float*)smem;
    int*   inds = (int*)(smem + TOPK_CAP * sizeof(float));
    __shared__ int   wcnt[8];
    __shared__ float redV[256];
    __shared__ int   redS[256];

    const int b = blockIdx.x / KC;
    const int k = blockIdx.x % KC;
    const int tid = threadIdx.x;
    const int wave = tid >> 5, lane = tid & 31;
    const float* row = sims + (size_t)b * NN;

    // --- deterministic stream compaction of cluster-k items
    int total = 0;
    for (int start = 0; start < NN; start += 256) {
        int n = start + tid;
        bool pred = (n < NN) && (labels[n] == k);
        u32 wm = (u32)__ballot(pred);
        if (lane == 0) wcnt[wave] = __popc(wm);
        __syncthreads();
        int wbase = total, chunkTotal = 0;
        for (int w = 0; w < 8; ++w) {
            int c = wcnt[w];
            if (w < wave) wbase += c;
            chunkTotal += c;
        }
        if (pred) {
            int pos = wbase + __popc(wm & ((1u << lane) - 1u));
            if (pos < TOPK_CAP) { vals[pos] = row[n]; inds[pos] = n; }
        }
        total += chunkTotal;
        __syncthreads();
    }
    int count = total < TOPK_CAP ? total : TOPK_CAP;

    // --- R rounds of argmax (ties -> smaller item index), knock out winner
    for (int r = 0; r < RR; ++r) {
        float bv = -INFINITY;
        int   bs = -1;
        for (int i = tid; i < count; i += 256) {
            float v = vals[i];
            if (v > bv || (v == bv && bs >= 0 && inds[i] < inds[bs])) { bv = v; bs = i; }
        }
        redV[tid] = bv; redS[tid] = bs;
        __syncthreads();
        for (int s = 128; s > 0; s >>= 1) {
            if (tid < s) {
                float v2 = redV[tid + s]; int s2 = redS[tid + s];
                bool take = (s2 >= 0) &&
                            (v2 > redV[tid] ||
                             (v2 == redV[tid] && (redS[tid] < 0 || inds[s2] < inds[redS[tid]])));
                if (take) { redV[tid] = v2; redS[tid] = s2; }
            }
            __syncthreads();
        }
        if (tid == 0) {
            int slot = redS[0];
            outIdx[((size_t)b * KC + k) * RR + r] = (slot >= 0) ? inds[slot] : 0;
            if (slot >= 0) vals[slot] = -INFINITY;
        }
        __syncthreads();
    }
}

// ---------------------------------------------------------------------------
// Gather feat[b*R + r, :] = [ u[b] | it[idx[b,0,r]] | ... | it[idx[b,15,r]] ]
// ---------------------------------------------------------------------------
__global__ void gather_feat(const float* __restrict__ u,
                            const u16* __restrict__ itB,
                            const int* __restrict__ idx,
                            u16* __restrict__ feat) {
    const int rowId = blockIdx.x;            // [0, B*R)
    const int b = rowId / RR;
    const int r = rowId % RR;
    const int W = HH * (KC + 1);
    for (int c = threadIdx.x; c < W; c += blockDim.x) {
        u16 v;
        if (c < HH) {
            v = f32_to_bf16(u[(size_t)b * HH + c]);
        } else {
            int k  = (c - HH) >> 9;          // /512
            int cc = (c - HH) & 511;
            int id = idx[((size_t)b * KC + k) * RR + r];
            v = itB[(size_t)id * HH + cc];
        }
        feat[(size_t)rowId * W + c] = v;
    }
}

// ---------------------------------------------------------------------------
// Host-side orchestration
// ---------------------------------------------------------------------------
extern "C" void kernel_launch(void* const* d_in, const int* in_sizes, int n_in,
                              void* d_out, int out_size, void* d_ws, size_t ws_size,
                              hipStream_t stream) {
    (void)in_sizes; (void)n_in; (void)out_size; (void)ws_size;

    const float* user       = (const float*)d_in[0];
    const float* items      = (const float*)d_in[1];
    const int*   hist       = (const int*)  d_in[2];
    const int*   item_lab   = (const int*)  d_in[3];
    const float* prompt_emb = (const float*)d_in[4];
    const float* uw1 = (const float*)d_in[5];  const float* ub1 = (const float*)d_in[6];
    const float* uw2 = (const float*)d_in[7];  const float* ub2 = (const float*)d_in[8];
    const float* iw1 = (const float*)d_in[9];  const float* ib1 = (const float*)d_in[10];
    const float* iw2 = (const float*)d_in[11]; const float* ib2 = (const float*)d_in[12];
    const float* tw1 = (const float*)d_in[13]; const float* tb1 = (const float*)d_in[14];
    const float* tw2 = (const float*)d_in[15]; const float* tb2 = (const float*)d_in[16];
    float* out = (float*)d_out;

    const int UK = SS * 2 * DD;   // 12800
    const int TK = HH * (KC + 1); // 8704
    const int BR = BB * RR;       // 3200

    // bump allocator over d_ws (256B aligned)
    size_t off = 0;
    char* base = (char*)d_ws;
    auto alloc = [&](size_t bytes) -> void* {
        void* p = base + off;
        off = (off + bytes + 255) & ~(size_t)255;
        return p;
    };
    u16*   itemsB    = (u16*)  alloc((size_t)NN * DD * 2);
    u16*   uw1T      = (u16*)  alloc((size_t)UK * HH * 2);   // [H, UK]
    u16*   uw2T      = (u16*)  alloc((size_t)HH * HH * 2);   // [H, H]
    u16*   iw1T      = (u16*)  alloc((size_t)DD * HH * 2);   // [H, D]
    u16*   iw2T      = (u16*)  alloc((size_t)HH * HH * 2);   // [H, H]
    u16*   tw1T      = (u16*)  alloc((size_t)TK * HH * 2);   // [H, TK]
    u16*   tw2T      = (u16*)  alloc((size_t)HH * KC * 2);   // [KC, H]
    u16*   promptedB = (u16*)  alloc((size_t)BB * UK * 2);
    u16*   huB       = (u16*)  alloc((size_t)BB * HH * 2);
    float* uF        = (float*)alloc((size_t)BB * HH * 4);
    u16*   uB        = (u16*)  alloc((size_t)BB * HH * 2);
    float* invNu     = (float*)alloc((size_t)BB * 4);
    u16*   h1B       = (u16*)  alloc((size_t)NN * HH * 2);   // reused later as featB
    u16*   itB       = (u16*)  alloc((size_t)NN * HH * 2);
    float* invNit    = (float*)alloc((size_t)NN * 4);
    float* sims      = (float*)alloc((size_t)BB * NN * 4);
    int*   idxBuf    = (int*)  alloc((size_t)BB * KC * RR * 4);
    u16*   featB     = h1B;      // h1 dead after item layer 2
    u16*   htB       = itemsB;   // items_bf16 dead after item layer 1

    auto cvt = [&](const float* x, u16* y, long n) {
        int g = (int)((n + 255) / 256); if (g > 4096) g = 4096;
        cvt_f32_bf16<<<g, 256, 0, stream>>>(x, y, (int)n);
    };
    auto cvtT = [&](const float* W, u16* Wt, int K, int N) {
        dim3 g((N + 31) / 32, (K + 31) / 32);
        cvt_transpose_f32_bf16<<<g, dim3(32, 8), 0, stream>>>(W, Wt, K, N);
    };
    auto grid_for = [](int M, int N) { return dim3((N + BN - 1) / BN, (M + BM - 1) / BM); };

    // ---- bf16 conversions: A matrices plain, weights transposed to [N,K]
    cvt(items, itemsB, (long)NN * DD);
    cvtT(uw1, uw1T, UK, HH);
    cvtT(uw2, uw2T, HH, HH);
    cvtT(iw1, iw1T, DD, HH);
    cvtT(iw2, iw2T, HH, HH);
    cvtT(tw1, tw1T, TK, HH);
    cvtT(tw2, tw2T, HH, KC);

    // ---- user tower
    build_prompted<<<BB, 256, 0, stream>>>(user, prompt_emb, hist, promptedB);
    wmma_gemm<true, false, true, true><<<grid_for(BB, HH), 256, 0, stream>>>(
        promptedB, uw1T, ub1, nullptr, nullptr, huB, BB, HH, UK);
    wmma_gemm<false, false, true, false><<<grid_for(BB, HH), 256, 0, stream>>>(
        huB, uw2T, ub2, nullptr, nullptr, uF, BB, HH, HH);
    cvt(uF, uB, (long)BB * HH);

    // ---- item tower (dominant GEMMs)
    wmma_gemm<true, false, true, true><<<grid_for(NN, HH), 256, 0, stream>>>(
        itemsB, iw1T, ib1, nullptr, nullptr, h1B, NN, HH, DD);
    wmma_gemm<true, false, true, false><<<grid_for(NN, HH), 256, 0, stream>>>(
        h1B, iw2T, ib2, nullptr, nullptr, itB, NN, HH, HH);

    // ---- norms + cosine sims (scales folded into GEMM epilogue)
    row_inv_norm<<<(BB + 7) / 8, 256, 0, stream>>>(uB, invNu, BB);
    row_inv_norm<<<(NN + 7) / 8, 256, 0, stream>>>(itB, invNit, NN);
    wmma_gemm<false, true, false, false><<<grid_for(BB, NN), 256, 0, stream>>>(
        uB, itB, nullptr, invNu, invNit, sims, BB, NN, HH);

    // ---- per-cluster deterministic top-R
    topk_per_cluster<<<BB * KC, 256, TOPK_CAP * 8, stream>>>(sims, item_lab, idxBuf);

    // ---- feature gather + task MLP
    gather_feat<<<BR, 256, 0, stream>>>(uF, itB, idxBuf, featB);
    wmma_gemm<true, false, true, true><<<grid_for(BR, HH), 256, 0, stream>>>(
        featB, tw1T, tb1, nullptr, nullptr, htB, BR, HH, TK);
    wmma_gemm<false, false, true, false><<<grid_for(BR, KC), 256, 0, stream>>>(
        htB, tw2T, tb2, nullptr, nullptr, out, BR, KC, HH);
}
